// PromptPool_42228118454565
// MI455X (gfx1250) — matrix operand order repaired and checked by reading, not compiled, not run on hardware
//
#include <hip/hip_runtime.h>
#include <hip/hip_bf16.h>
#include <stdint.h>

// Problem constants (match reference)
#define BQ    4096
#define POOL  1024
#define DIM   768
#define TOPK  5
#define PLEN  5
#define ROWF  (PLEN * DIM)      // 3840 floats per gathered row
#define ROWV4 (ROWF / 4)        // 960 float4 per row

typedef __bf16 bf16;
typedef bf16  v16bf __attribute__((ext_vector_type(16)));
typedef bf16  v8bf  __attribute__((ext_vector_type(8)));
typedef float v8f   __attribute__((ext_vector_type(8)));
typedef float v4f   __attribute__((ext_vector_type(4)));

// ---------------------------------------------------------------------------
// Kernel 1: L2-normalize rows of a [nrows, 768] f32 matrix, emit bf16 (RNE).
// ---------------------------------------------------------------------------
__global__ __launch_bounds__(256) void normalize_rows_bf16(
    const float* __restrict__ src, uint16_t* __restrict__ dst, int nrows) {
  int row = blockIdx.x;
  if (row >= nrows) return;
  const float* s = src + (size_t)row * DIM;

  float ss = 0.f;
  for (int i = threadIdx.x; i < DIM; i += 256) {
    float v = s[i];
    ss += v * v;
  }
  __shared__ float red[256];
  red[threadIdx.x] = ss;
  __syncthreads();
  for (int off = 128; off; off >>= 1) {
    if (threadIdx.x < off) red[threadIdx.x] += red[threadIdx.x + off];
    __syncthreads();
  }
  // matches F.normalize: x / max(||x||, eps)
  float scale = 1.0f / fmaxf(sqrtf(red[0]), 1e-12f);

  uint16_t* d = dst + (size_t)row * DIM;
  for (int i = threadIdx.x; i < DIM; i += 256) {
    float v = s[i] * scale;
    uint32_t u = __float_as_uint(v);
    uint32_t r = (u + 0x7FFFu + ((u >> 16) & 1u)) >> 16;  // RNE to bf16
    d[i] = (uint16_t)r;
  }
}

// ---------------------------------------------------------------------------
// Kernel 2: fused  distance-GEMM (WMMA bf16) + top-5 + gather.
// Block = 256 threads = 8 waves; block owns 16 queries, each wave owns a
// 128-wide pool slice for the GEMM, then 2 queries for top-k + gather.
// ---------------------------------------------------------------------------
__global__ __launch_bounds__(256) void prompt_pool_main(
    const uint16_t* __restrict__ qnb,   // [BQ, DIM] bf16 normalized queries
    const uint16_t* __restrict__ pkb,   // [POOL, DIM] bf16 normalized keys
    const float*    __restrict__ pv,    // [POOL, PLEN, DIM] f32 prompt values
    float*          __restrict__ out,   // [BQ, TOPK, PLEN, DIM] f32
    float*          __restrict__ partials) {  // [gridDim.x * 8] loss partials
  __shared__ float dist[16][POOL];      // exactly 64 KB

  const int tid  = threadIdx.x;
  const int wave = tid >> 5;
  const int lane = tid & 31;
  const int qm   = blockIdx.x * 16;     // first query of this block
  const int lrow  = lane & 15;
  const int lhalf = (lane >> 4) & 1;

  // ---- Phase 1: WMMA tiles, D = Q_tile(16xK) * K_tile(Kx16)^T ------------
  // A 16x32 bf16 per-lane layout: lanes 0-15 hold K 0..15 of row M,
  // lanes 16-31 hold K 16..31 (contiguous halves, matching B).
  // B 32x16 bf16 per-lane layout: lanes 0-15 hold K 0..15 of column N,
  // lanes 16-31 hold K 16..31 (contiguous halves).
  const uint16_t* aRow  = qnb + (size_t)(qm + lrow) * DIM;
  const int aOff = lhalf ? 16 : 0;      // A: elems0..15 <- k0+aOff
  const int bOff = lhalf ? 16 : 0;      // B: elems0..15 <- k0+bOff

  for (int ntp = 0; ntp < 4; ++ntp) {
    const int n0 = wave * 128 + ntp * 32;               // two N-tiles: n0, n0+16
    const uint16_t* bRow0 = pkb + (size_t)(n0 + lrow) * DIM;
    const uint16_t* bRow1 = pkb + (size_t)(n0 + 16 + lrow) * DIM;
    v8f acc0 = {};
    v8f acc1 = {};
    for (int k0 = 0; k0 < DIM; k0 += 32) {
      v8bf alo = *(const v8bf*)(aRow + k0 + aOff);
      v8bf ahi = *(const v8bf*)(aRow + k0 + aOff + 8);
      v8bf b0l = *(const v8bf*)(bRow0 + k0 + bOff);
      v8bf b0h = *(const v8bf*)(bRow0 + k0 + bOff + 8);
      v8bf b1l = *(const v8bf*)(bRow1 + k0 + bOff);
      v8bf b1h = *(const v8bf*)(bRow1 + k0 + bOff + 8);
      v16bf a  = __builtin_shufflevector(alo, ahi, 0,1,2,3,4,5,6,7,8,9,10,11,12,13,14,15);
      v16bf b0 = __builtin_shufflevector(b0l, b0h, 0,1,2,3,4,5,6,7,8,9,10,11,12,13,14,15);
      v16bf b1 = __builtin_shufflevector(b1l, b1h, 0,1,2,3,4,5,6,7,8,9,10,11,12,13,14,15);
      acc0 = __builtin_amdgcn_wmma_f32_16x16x32_bf16(false, a, false, b0,
                                                     (short)0, acc0, false, false);
      acc1 = __builtin_amdgcn_wmma_f32_16x16x32_bf16(false, a, false, b1,
                                                     (short)0, acc1, false, false);
    }
    // C/D f32 layout: lane 0-15 -> N=lane, M=r; lane 16-31 -> N=lane-16, M=r+8
    const int M = lhalf ? 8 : 0;
#pragma unroll
    for (int r = 0; r < 8; ++r) {
      dist[M + r][n0 + lrow]      = acc0[r];
      dist[M + r][n0 + 16 + lrow] = acc1[r];
    }
  }
  __syncthreads();

  // ---- Phase 2: top-5 per query (wave handles queries 2w, 2w+1) ----------
  float wsum = 0.f;
  int idx0[TOPK], idx1[TOPK];           // wave-uniform after reduction
#pragma unroll
  for (int qq = 0; qq < 2; ++qq) {
    const int m = wave * 2 + qq;
    float* row = dist[m];
#pragma unroll
    for (int k = 0; k < TOPK; ++k) {
      float best = -INFINITY;
      int   bi   = POOL;
      for (int j = lane; j < POOL; j += 32) {
        float v = row[j];
        if (v > best) { best = v; bi = j; }     // strict > keeps lowest index
      }
#pragma unroll
      for (int off = 16; off; off >>= 1) {      // wave32 argmax reduce
        float ov = __shfl_xor(best, off, 32);
        int   oi = __shfl_xor(bi, off, 32);
        if (ov > best || (ov == best && oi < bi)) { best = ov; bi = oi; }
      }
      if (lane == (bi & 31)) row[bi] = -INFINITY;  // knock out for next round
      if (qq == 0) idx0[k] = bi; else idx1[k] = bi;
      wsum += fabsf(best);                    // same value on all lanes
    }
  }
  if (lane == 0) partials[blockIdx.x * 8 + wave] = wsum;

  // ---- Phase 3: gather prompt_values rows (per-wave, no barrier needed) --
#pragma unroll
  for (int qq = 0; qq < 2; ++qq) {
    const size_t b = (size_t)(qm + wave * 2 + qq);
#pragma unroll
    for (int k = 0; k < TOPK; ++k) {
      const int p = (qq == 0) ? idx0[k] : idx1[k];
      const v4f* __restrict__ sp = (const v4f*)(pv + (size_t)p * ROWF);
      v4f* __restrict__ dp = (v4f*)(out + (b * TOPK + k) * ROWF);
      for (int off = lane; off < ROWV4; off += 32) {
        __builtin_nontemporal_store(sp[off], dp + off);  // stream 315MB output
      }
    }
  }
}

// ---------------------------------------------------------------------------
// Kernel 3: deterministic fixed-order loss reduction over 2048 partials.
// ---------------------------------------------------------------------------
__global__ __launch_bounds__(256) void loss_reduce(
    const float* __restrict__ partials, float* __restrict__ out_loss) {
  __shared__ float red[256];
  float s = 0.f;
#pragma unroll
  for (int i = 0; i < 8; ++i) s += partials[threadIdx.x + 256 * i];
  red[threadIdx.x] = s;
  __syncthreads();
  for (int off = 128; off; off >>= 1) {
    if (threadIdx.x < off) red[threadIdx.x] += red[threadIdx.x + off];
    __syncthreads();
  }
  if (threadIdx.x == 0) out_loss[0] = red[0] / (float)BQ;
}

// ---------------------------------------------------------------------------
extern "C" void kernel_launch(void* const* d_in, const int* in_sizes, int n_in,
                              void* d_out, int out_size, void* d_ws, size_t ws_size,
                              hipStream_t stream) {
  const float* query = (const float*)d_in[0];   // [4096, 768]
  const float* pkeys = (const float*)d_in[1];   // [1024, 768]
  const float* pvals = (const float*)d_in[2];   // [1024, 5, 768]
  float* out = (float*)d_out;                   // 78643200 gathered + 1 loss

  // workspace layout: pk bf16 | qn bf16 | loss partials
  uint16_t* pkb = (uint16_t*)d_ws;
  uint16_t* qnb = (uint16_t*)((char*)d_ws + (size_t)POOL * DIM * 2);
  float* partials = (float*)((char*)d_ws + (size_t)POOL * DIM * 2
                                         + (size_t)BQ * DIM * 2);

  normalize_rows_bf16<<<POOL, 256, 0, stream>>>(pkeys, pkb, POOL);
  normalize_rows_bf16<<<BQ,   256, 0, stream>>>(query, qnb, BQ);
  prompt_pool_main<<<BQ / 16, 256, 0, stream>>>(qnb, pkb, pvals, out, partials);
  loss_reduce<<<1, 256, 0, stream>>>(partials,
                                     out + (size_t)BQ * TOPK * PLEN * DIM);
}
